// MeanMaxPooling_30296699306389
// MI455X (gfx1250) — matrix operand order repaired and checked by reading, not compiled, not run on hardware
//
#include <hip/hip_runtime.h>

// Problem dims (fixed by the reference)
#define N_  4
#define E_  64
#define L_  512
#define D_  768
#define K_  (2 * D_)        // 1536 (concat of max|mean)
#define M_  (N_ * E_)       // 256 GEMM rows
#define APAD   8            // LDS row padding (bf16 elems) -> conflict-free ds_load_b128
#define ALDST  (K_ + APAD)  // 1544

typedef unsigned short u16;
typedef __attribute__((ext_vector_type(16))) __bf16 v16bf;
typedef __attribute__((ext_vector_type(8)))  float  v8f;

union Frag16 { v16bf v; uint4 u[2]; };

__device__ __forceinline__ u16 f2bf(float f) {
  unsigned int u = __float_as_uint(f);
  u += 0x7FFFu + ((u >> 16) & 1u);       // round-to-nearest-even
  return (u16)(u >> 16);
}

// ---------------------------------------------------------------------------
// Kernel 1: convert W [D, 2D] fp32 -> bf16 (L2-resident, one pass)
// ---------------------------------------------------------------------------
__global__ void convert_w_kernel(const float* __restrict__ W,
                                 u16* __restrict__ Wbf, int n) {
  int i = blockIdx.x * blockDim.x + threadIdx.x;
  const int stride = gridDim.x * blockDim.x;
  for (; i < n; i += stride) Wbf[i] = f2bf(W[i]);
}

// ---------------------------------------------------------------------------
// Kernel 2: fused masked max+mean pooling, writing the GEMM A-matrix
// directly in bf16 as [M_, K_] = [max | mean].
// Each block: one n, a group of 8 entities, 256 consecutive d-columns.
// One doc_state read feeds 8 entity accumulator pairs -> 8x less L2 traffic
// than the naive one-block-per-entity mapping (~50 MB instead of ~400 MB).
// ---------------------------------------------------------------------------
__global__ __launch_bounds__(256) void pool_kernel(
    const float* __restrict__ doc,   // [N, L, D]
    const float* __restrict__ map,   // [N, E, L]
    const float* __restrict__ lens,  // [N, E]
    u16* __restrict__ Abf) {         // [M_, K_] bf16
  const int dchunk = blockIdx.x;          // 0..2
  const int eg     = blockIdx.y;          // 0..7  (entities eg*8 .. eg*8+7)
  const int n      = blockIdx.z;          // 0..3
  const int e0     = eg * 8;
  const int d      = dchunk * 256 + threadIdx.x;

  __shared__ float smask[8 * L_];         // 16 KB: mask rows of 8 entities
  for (int i = threadIdx.x; i < 8 * L_; i += 256) {
    const int j = i >> 9;                 // entity within group
    const int l = i & (L_ - 1);
    smask[i] = map[((size_t)(n * E_ + e0 + j)) * L_ + l];
  }
  __syncthreads();

  float mx[8], sm[8];
  #pragma unroll
  for (int j = 0; j < 8; ++j) { mx[j] = -__builtin_inff(); sm[j] = 0.f; }

  const float* base = doc + (size_t)n * L_ * D_ + d;
  for (int l = 0; l < L_; ++l) {
    const float v = base[(size_t)l * D_];       // one doc read feeds 8 entities
    #pragma unroll
    for (int j = 0; j < 8; ++j) {
      const float p = smask[j * L_ + l] * v;    // LDS broadcast, no conflicts
      mx[j] = fmaxf(mx[j], p);                  // zeros participate, matching ref
      sm[j] += p;
    }
  }

  #pragma unroll
  for (int j = 0; j < 8; ++j) {
    const int ne = n * E_ + e0 + j;
    const float inv = 1.0f / lens[ne];
    u16* outr = Abf + (size_t)ne * K_;
    outr[d]      = f2bf(mx[j]);
    outr[D_ + d] = f2bf(sm[j] * inv);
  }
}

// ---------------------------------------------------------------------------
// Kernel 3: WMMA GEMM  out[m, c] = sum_k A[m,k] * W[c,k] + b[c]
// M=256, N=768, K=1536, bf16 x bf16 -> fp32 via v_wmma_f32_16x16x32_bf16.
// Block = 256 threads = 8 wave32; each wave owns TWO adjacent 16x16 output
// tiles (cols wave*32 and wave*32+16 within a 256-col block): the A fragment
// is loaded once per K-step and reused by two independent WMMA accumulator
// chains. A row-tile staged in padded LDS once per block, shared by 8 waves.
// ---------------------------------------------------------------------------
__global__ __launch_bounds__(256) void gemm_wmma_kernel(
    const u16* __restrict__ Abf,    // [M_, K_] bf16
    const u16* __restrict__ Wbf,    // [D_, K_] bf16 (K-contiguous)
    const float* __restrict__ bias, // [D_]
    float* __restrict__ out) {      // [M_, D_]
  __shared__ u16 sA[16 * ALDST];    // 48.25 KB
  const int tid  = threadIdx.x;
  const int wave = tid >> 5;
  const int lane = tid & 31;
  const int row0 = blockIdx.y * 16;
  const int col0 = blockIdx.x * 256 + wave * 32;   // first of two 16-col tiles

  // Cooperative stage: 16 rows x 1536 bf16, 16B chunks, padded rows.
  for (int i = tid; i < 16 * (K_ / 8); i += 256) {
    const int r = i / (K_ / 8);
    const int c = i % (K_ / 8);
    *(uint4*)(&sA[r * ALDST + c * 8]) =
        *(const uint4*)(Abf + (size_t)(row0 + r) * K_ + c * 8);
  }
  __syncthreads();

  const int nn    = lane & 15;     // A row within tile / B column within tile
  const int khalf = lane >> 4;     // which K half this lane holds
  // A 16-bit layout (ISA 7.12.2): lane half khalf holds K = 8*khalf + {0..7}
  // in elems 0..7 and K = 16 + 8*khalf + {0..7} in elems 8..15.
  const u16* aRow  = &sA[nn * ALDST + khalf * 8];
  // B 16-bit layout (matches A): lane half khalf holds K = 8*khalf + {0..7}
  // in elems 0..7 and K = 16 + 8*khalf + {0..7} in elems 8..15. B[k, c] = W[c, k] -> read W row.
  const u16* bRow0 = Wbf + (size_t)(col0 + nn) * K_ + khalf * 8;
  const u16* bRow1 = bRow0 + (size_t)16 * K_;          // second tile's columns

  v8f acc0 = {}, acc1 = {};
  #pragma unroll 4
  for (int k0 = 0; k0 < K_; k0 += 32) {
    Frag16 fa, fb0, fb1;
    fa.u[0]  = *(const uint4*)(aRow + k0);         // ds_load_b128
    fa.u[1]  = *(const uint4*)(aRow + k0 + 16);    // ds_load_b128
    fb0.u[0] = *(const uint4*)(bRow0 + k0);        // global_load_b128 (L2 hit)
    fb0.u[1] = *(const uint4*)(bRow0 + k0 + 16);
    fb1.u[0] = *(const uint4*)(bRow1 + k0);
    fb1.u[1] = *(const uint4*)(bRow1 + k0 + 16);
    __builtin_prefetch(bRow0 + k0 + 128, 0, 3);    // WGP-scope prefetch of B
    __builtin_prefetch(bRow1 + k0 + 128, 0, 3);
    acc0 = __builtin_amdgcn_wmma_f32_16x16x32_bf16(
        false, fa.v, false, fb0.v, (short)0, acc0, false, false);
    acc1 = __builtin_amdgcn_wmma_f32_16x16x32_bf16(
        false, fa.v, false, fb1.v, (short)0, acc1, false, false);
  }

  // C/D layout: lanes 0-15 -> M = vgpr, lanes 16-31 -> M = vgpr + 8; N = lane&15
  const float bv0 = bias[col0 + nn];
  const float bv1 = bias[col0 + 16 + nn];
  #pragma unroll
  for (int i = 0; i < 8; ++i) {
    const int m = row0 + i + 8 * khalf;
    out[(size_t)m * D_ + col0 + nn]      = acc0[i] + bv0;
    out[(size_t)m * D_ + col0 + 16 + nn] = acc1[i] + bv1;
  }
}

// ---------------------------------------------------------------------------
extern "C" void kernel_launch(void* const* d_in, const int* in_sizes, int n_in,
                              void* d_out, int out_size, void* d_ws, size_t ws_size,
                              hipStream_t stream) {
  (void)in_sizes; (void)n_in; (void)out_size; (void)ws_size;
  const float* doc  = (const float*)d_in[0];  // [N, L, D]
  const float* map  = (const float*)d_in[1];  // [N, E, L]
  const float* lens = (const float*)d_in[2];  // [N, E]
  const float* W    = (const float*)d_in[3];  // [D, 2D]
  const float* b    = (const float*)d_in[4];  // [D]
  float* out = (float*)d_out;                 // [N, E, D]

  // Workspace: Wbf (D_*K_ bf16 = 2.25 MB) then Abf (M_*K_ bf16 = 0.75 MB)
  u16* Wbf = (u16*)d_ws;
  u16* Abf = (u16*)((char*)d_ws + (size_t)D_ * K_ * sizeof(u16));

  convert_w_kernel<<<2304, 256, 0, stream>>>(W, Wbf, D_ * K_);

  dim3 pgrid(D_ / 256, E_ / 8, N_);           // (3, 8, 4)
  pool_kernel<<<pgrid, 256, 0, stream>>>(doc, map, lens, Abf);

  dim3 ggrid(D_ / 256, M_ / 16);              // (3, 16), 8 waves x 2 tiles
  gemm_wmma_kernel<<<ggrid, 256, 0, stream>>>(Abf, Wbf, b, out);
}
